// SRCapsNet_31817117729385
// MI455X (gfx1250) — matrix operand order repaired and checked by reading, not compiled, not run on hardware
//
#include <hip/hip_runtime.h>
#include <hip/hip_bf16.h>

typedef __attribute__((ext_vector_type(16))) __bf16 v16bf;
typedef __attribute__((ext_vector_type(8)))  float  v8f;
typedef __attribute__((ext_vector_type(4)))  unsigned int v4u;
typedef __attribute__((ext_vector_type(8)))  int v8i;
typedef __attribute__((ext_vector_type(4)))  int v4i;

#define D  256
#define B  128
#define LW 64
#define NS 32
#define NI 16

// pack two f32 -> two bf16 (truncate) in ONE v_perm_b32:
// out[15:0] = lo[31:16], out[31:16] = hi[31:16]
__device__ inline unsigned int pack2(float lo, float hi) {
  return __builtin_amdgcn_perm(__float_as_uint(hi), __float_as_uint(lo), 0x07060302u);
}

// ---- TDM: DMA a 32-row x 64-col f32 tile (row stride 256 floats) into LDS ----
__device__ inline void tdm_load_tile(const float* gaddr, unsigned lds_off) {
  unsigned long long ga = (unsigned long long)(uintptr_t)gaddr;
  v4u g0;
  g0[0] = 1u;                                        // count=1, is_restore=0, gather off
  g0[1] = lds_off;                                   // lds_addr [63:32]
  g0[2] = (unsigned)(ga & 0xFFFFFFFFu);              // global_addr lo
  g0[3] = (unsigned)((ga >> 32) & 0x01FFFFFFu)       // global_addr hi [120:96]
          | (2u << 30);                              // type=2 (image)
  v8i g1;
  g1[0] = (int)(2u << 16);                           // data_size=2 -> 4 bytes
  g1[1] = (int)(256u << 16);                         // tensor_dim0 = 256 (bits 48..63)
  g1[2] = (int)(256u << 16);                         // tensor_dim1 = 256 (bits 80..95)
  g1[3] = (int)(64u << 16);                          // tile_dim0 = 64 (bits 112..127)
  g1[4] = 32;                                        // tile_dim1 = 32
  g1[5] = 256;                                       // tensor_dim0_stride = 256
  g1[6] = 0;
  g1[7] = 0;
  v4i z4 = {0, 0, 0, 0};
  v8i z8 = {0, 0, 0, 0, 0, 0, 0, 0};
  __builtin_amdgcn_tensor_load_to_lds(g0, g1, z4, z4, z8, 0);
}

// ---------------- zero scratch accumulators ----------------
__global__ void zero_kernel(float* p, int n) {
  int i = blockIdx.x * blockDim.x + threadIdx.x;
  if (i < n) p[i] = 0.0f;
}

// ---------------- words -> slots routing (softmax), per-(b,l) wave ----------------
__global__ void route_words_kernel(const float* __restrict__ X, const float* __restrict__ Wr,
                                   float* __restrict__ R, float* __restrict__ WCs,
                                   float* __restrict__ Asum) {
  __shared__ float xsh[8][D];
  int wave = threadIdx.x >> 5, lane = threadIdx.x & 31;
  int id = blockIdx.x * 8 + wave;            // 0..8191 = b*64 + l
  int b = id >> 6, l = id & 63;
  const float* xp = X + (size_t)(b * LW + l) * D;
  float part = 0.0f;
#pragma unroll
  for (int t = 0; t < 8; ++t) { float v = xp[lane * 8 + t]; xsh[wave][lane * 8 + t] = v; part += v; }
  __syncthreads();
  for (int m = 16; m > 0; m >>= 1) part += __shfl_xor(part, m, 32);
  float a_w = part * (1.0f / D);             // a_words[b,l]
  const float* wr = Wr + (size_t)(l * NS + lane) * D;
  float s = 0.0f;
  for (int j = 0; j < D; ++j) s += wr[j] * xsh[wave][j];
  float mx = s;
  for (int m = 16; m > 0; m >>= 1) mx = fmaxf(mx, __shfl_xor(mx, m, 32));
  float e = __expf(s - mx);
  float se = e;
  for (int m = 16; m > 0; m >>= 1) se += __shfl_xor(se, m, 32);
  float r = (e / se) * a_w;                  // c_ws * a_words
  R[(size_t)(b * LW + l) * NS + lane] = r;
  atomicAdd(&WCs[b * NS + lane], r);         // weighted_c[b,s]
  if (lane == 0) atomicAdd(&Asum[b], a_w);   // sum_l a_words
}

// ---------------- a_slots = wc / sum(a_words); Asum2 = sum_s a_slots ----------------
__global__ void finalize_slots_kernel(const float* __restrict__ WCs, const float* __restrict__ Asum,
                                      float* __restrict__ a_slots_out, float* __restrict__ Asum2) {
  int b = blockIdx.x, s = threadIdx.x;
  float as = WCs[b * NS + s] / Asum[b];
  a_slots_out[b * NS + s] = as;
  float t = as;
  for (int m = 16; m > 0; m >>= 1) t += __shfl_xor(t, m, 32);
  if (s == 0) Asum2[b] = t;
}

// ---------------- u_slots: fused routed GEMM, bf16 WMMA + TDM double buffer ----------------
// u_slots[b,s,k] = (1/wc[b,s]) * sum_{l,j} r[b,l,s] * Wp[l,s,k,j] * x[b,l,j]
// block = (s, ky); 8 waves; wave w -> b rows [16w,16w+16), two 16x16 N-tiles.
// Stage = (l, 64-wide j chunk): TDM streams the f32 weight tile, threads stage r-scaled
// bf16 x; 4 WMMAs per stage per wave.
__global__ void __launch_bounds__(256)
uslots_wmma_kernel(const float* __restrict__ X, const float* __restrict__ Wp,
                   const float* __restrict__ R, const float* __restrict__ WCs,
                   float* __restrict__ Uout) {
  __shared__ __align__(16) unsigned short xs[2][B][64];   // r-scaled x, bf16, 2x16KB
  __shared__ __align__(16) float wt[2][32][64];           // raw f32 weight tiles, 2x8KB
  const int o = blockIdx.x;                               // slot s
  const int ky = blockIdx.y;                              // 32-wide k tile
  const int tid = threadIdx.x, wave = tid >> 5, lane = tid & 31;
  const int lm = lane & 15, h = lane >> 4;
  const int T = LW * 4;                                   // 256 stages (l, jc)
  v8f acc0 = {}; v8f acc1 = {};

  // ---- prologue: stage 0 ----
  {
    int row = tid >> 1, c0 = (tid & 1) * 32;
    float rv = R[(size_t)(row * LW + 0) * NS + o];
    const float4* xp4 = (const float4*)(X + (size_t)(row * LW + 0) * D + c0);
    unsigned int* dst = (unsigned int*)&xs[0][row][c0];
#pragma unroll
    for (int i = 0; i < 8; ++i) {
      float4 f = xp4[i];
      dst[2 * i + 0] = pack2(f.x * rv, f.y * rv);
      dst[2 * i + 1] = pack2(f.z * rv, f.w * rv);
    }
  }
  if (wave == 0) {
    const float* g = Wp + ((((size_t)0 * NS + o) * D) + (size_t)(ky * 32)) * D;
    tdm_load_tile(g, (unsigned)(uintptr_t)&wt[0][0][0]);
  }

  for (int t = 0; t < T; ++t) {
    const int cur = t & 1, nxt = cur ^ 1;
    if (wave == 0) __builtin_amdgcn_s_wait_tensorcnt(0);  // weight tile t landed
    __syncthreads();                                      // publish tile t (x + W)
    if (t + 1 < T) {
      const int l1 = (t + 1) >> 2, jc1 = ((t + 1) & 3) * 64;
      if (wave == 0) {                                    // overlap: DMA tile t+1
        const float* g = Wp + ((((size_t)l1 * NS + o) * D) + (size_t)(ky * 32)) * D + jc1;
        tdm_load_tile(g, (unsigned)(uintptr_t)&wt[nxt][0][0]);
      }
      {                                                   // stage x tile t+1
        int row = tid >> 1, c0 = (tid & 1) * 32;
        float rv = R[(size_t)(row * LW + l1) * NS + o];
        const float4* xp4 = (const float4*)(X + (size_t)(row * LW + l1) * D + jc1 + c0);
        unsigned int* dst = (unsigned int*)&xs[nxt][row][c0];
#pragma unroll
        for (int i = 0; i < 8; ++i) {
          float4 f = xp4[i];
          dst[2 * i + 0] = pack2(f.x * rv, f.y * rv);
          dst[2 * i + 1] = pack2(f.z * rv, f.w * rv);
        }
      }
    }
    // ---- compute on tile t ----
    union F { uint4 u[2]; v16bf v; };
    F a[2];
    const uint4* xr = (const uint4*)&xs[cur][wave * 16 + lm][0];  // 128B row = 8 uint4
#pragma unroll
    for (int jt = 0; jt < 2; ++jt) {       // A layout: h=0 -> K{0..15}, h=1 -> K{16..31}
      a[jt].u[0] = xr[jt * 4 + 2 * h];
      a[jt].u[1] = xr[jt * 4 + 2 * h + 1];
    }
#pragma unroll
    for (int kt = 0; kt < 2; ++kt) {
#pragma unroll
      for (int jt = 0; jt < 2; ++jt) {
        const float* wrow = &wt[cur][kt * 16 + lm][jt * 32 + h * 16];  // lane: N fixed, 16 K's
        F bf;
#pragma unroll
        for (int q = 0; q < 8; ++q)
          ((unsigned int*)&bf)[q] = pack2(wrow[2 * q], wrow[2 * q + 1]);
        if (kt == 0)
          acc0 = __builtin_amdgcn_wmma_f32_16x16x32_bf16(false, a[jt].v, false, bf.v, (short)0, acc0, false, false);
        else
          acc1 = __builtin_amdgcn_wmma_f32_16x16x32_bf16(false, a[jt].v, false, bf.v, (short)0, acc1, false, false);
      }
    }
  }
  // epilogue: C layout -> row M = h*8+p, col N = lane&15; divide by weighted_c
#pragma unroll
  for (int p = 0; p < 8; ++p) {
    int brow = wave * 16 + h * 8 + p;
    float inv = __builtin_amdgcn_rcpf(WCs[brow * NS + o]);
    size_t base = (size_t)(brow * NS + o) * D + ky * 32 + lm;
    Uout[base]      = acc0[p] * inv;
    Uout[base + 16] = acc1[p] * inv;
  }
}

// ---------------- slots -> intents routing (softmax over 16), per-(b,s) wave ----------------
__global__ void route_slots_kernel(const float* __restrict__ Uslots, const float* __restrict__ Wr2,
                                   const float* __restrict__ a_slots, float* __restrict__ R2,
                                   float* __restrict__ WC2) {
  __shared__ float ush[8][D];
  int wave = threadIdx.x >> 5, lane = threadIdx.x & 31;
  int id = blockIdx.x * 8 + wave;            // 0..4095 = b*32 + s
  int b = id >> 5, s = id & 31;
  const float* up = Uslots + (size_t)(b * NS + s) * D;
#pragma unroll
  for (int t = 0; t < 8; ++t) ush[wave][lane * 8 + t] = up[lane * 8 + t];
  __syncthreads();
  int i = lane & 15, h = lane >> 4;
  const float* wr = Wr2 + (size_t)(s * NI + i) * D + h * 128;
  float part = 0.0f;
  for (int j = 0; j < 128; ++j) part += wr[j] * ush[wave][h * 128 + j];
  float tot = part + __shfl_xor(part, 16, 32);
  float mx = tot;
  for (int m = 8; m > 0; m >>= 1) mx = fmaxf(mx, __shfl_xor(mx, m, 32));
  float e = __expf(tot - mx);
  float se = e;
  for (int m = 8; m > 0; m >>= 1) se += __shfl_xor(se, m, 32);
  float r2 = (e / se) * a_slots[b * NS + s];
  if (h == 0) {
    R2[(size_t)(b * NS + s) * NI + i] = r2;
    atomicAdd(&WC2[b * NI + i], r2);
  }
}

__global__ void finalize_intents_kernel(const float* __restrict__ WC2, const float* __restrict__ Asum2,
                                        float* __restrict__ a_int_out) {
  int idx = blockIdx.x * blockDim.x + threadIdx.x;
  if (idx >= B * NI) return;
  a_int_out[idx] = WC2[idx] / Asum2[idx >> 4];
}

// ---------------- S[s,i,k] = sum_j Wp_si[s,i,j,k]  (the 'sijk,bsk' quirk) ----------------
__global__ void reduce_s_kernel(const float* __restrict__ Wp2, float* __restrict__ S) {
  int idx = blockIdx.x * blockDim.x + threadIdx.x;  // (s*16+i)*256 + k
  int k = idx & 255, si = idx >> 8;
  const float* base = Wp2 + (size_t)si * D * D + k;
  float acc = 0.0f;
  for (int j = 0; j < D; ++j) acc += base[(size_t)j * D];
  S[idx] = acc;
}

// u_intents[b,i,k] = (1/wc2[b,i]) * sum_s r2[b,s,i] * S[s,i,k] * u_slots[b,s,k]
__global__ void uintents_kernel(const float* __restrict__ S, const float* __restrict__ Uslots,
                                const float* __restrict__ R2, const float* __restrict__ WC2,
                                float* __restrict__ Uint) {
  int bi = blockIdx.x;                    // b*NI + i
  int b = bi >> 4, i = bi & 15;
  int k = threadIdx.x;
  float acc = 0.0f;
  for (int s = 0; s < NS; ++s) {
    float r = R2[(size_t)(b * NS + s) * NI + i];
    acc += r * S[(size_t)(s * NI + i) * D + k] * Uslots[(size_t)(b * NS + s) * D + k];
  }
  Uint[(size_t)bi * D + k] = acc / WC2[b * NI + i];
}

extern "C" void kernel_launch(void* const* d_in, const int* in_sizes, int n_in,
                              void* d_out, int out_size, void* d_ws, size_t ws_size,
                              hipStream_t stream) {
  const float* X   = (const float*)d_in[0];   // (128,64,256)
  const float* Wr  = (const float*)d_in[1];   // (64,32,256)
  const float* Wp  = (const float*)d_in[2];   // (64,32,256,256)
  const float* Wr2 = (const float*)d_in[3];   // (32,16,256)
  const float* Wp2 = (const float*)d_in[4];   // (32,16,256,256)

  float* out = (float*)d_out;
  float* a_slots   = out;                                  // 4096
  float* u_slots   = out + 4096;                           // 1048576
  float* a_intents = out + 4096 + 1048576;                 // 2048
  float* u_intents = out + 4096 + 1048576 + 2048;          // 524288

  float* ws    = (float*)d_ws;
  float* R     = ws;                 // 262144  r[b,l,s]
  float* WCs   = ws + 262144;        // 4096    weighted_c[b,s]
  float* Asum  = ws + 266240;        // 128     sum_l a_words
  float* Asum2 = ws + 266368;        // 128     sum_s a_slots
  float* WC2   = ws + 266496;        // 2048    weighted_c2[b,i]
  float* R2    = ws + 268544;        // 65536   r2[b,s,i]
  float* S     = ws + 334080;        // 131072  S[s,i,k]

  zero_kernel<<<(6400 + 255) / 256, 256, 0, stream>>>(WCs, 6400);  // WCs+Asum+Asum2+WC2
  route_words_kernel<<<1024, 256, 0, stream>>>(X, Wr, R, WCs, Asum);
  finalize_slots_kernel<<<128, 32, 0, stream>>>(WCs, Asum, a_slots, Asum2);
  uslots_wmma_kernel<<<dim3(NS, 8), 256, 0, stream>>>(X, Wp, R, WCs, u_slots);
  reduce_s_kernel<<<131072 / 256, 256, 0, stream>>>(Wp2, S);
  route_slots_kernel<<<512, 256, 0, stream>>>(u_slots, Wr2, a_slots, R2, WC2);
  finalize_intents_kernel<<<(B * NI + 63) / 64, 64, 0, stream>>>(WC2, Asum2, a_intents);
  uintents_kernel<<<B * NI, 256, 0, stream>>>(S, u_slots, R2, WC2, u_intents);
}
